// SparseLinear_31164282700431
// MI455X (gfx1250) — compile-verified, matrix operand order checked by reading, not run on hardware
//
#include <hip/hip_runtime.h>

// ---------------- CDNA5 vector types ----------------
typedef __attribute__((ext_vector_type(16))) __bf16 v16bf;
typedef __attribute__((ext_vector_type(8)))  __bf16 v8bf;
typedef __attribute__((ext_vector_type(8)))  float  v8f;

#define NROWS 16384
#define CIN   4096
#define COUT  4096

#define TM 128     // block tile rows of x
#define TN 128     // block tile cols of y (rows of W)
#define TK 32      // K chunk (bf16 wmma K)
#define LDT 40     // padded LDS row stride (elements) -> 80B, keeps ds b128 alignment

__device__ __forceinline__ unsigned short bf16_bits(float f) {
    return __builtin_bit_cast(unsigned short, (__bf16)f);   // native RNE f32->bf16
}

__global__ __launch_bounds__(256)
void sparse_linear_wmma_bf16x3(const float* __restrict__ x,
                               const float* __restrict__ w,
                               const float* __restrict__ bias,
                               float* __restrict__ out)
{
    // bf16 split tiles: A = x rows, B = W rows (columns of y). hi + lo parts.
    __shared__ __align__(16) unsigned short sAhi[TM * LDT];
    __shared__ __align__(16) unsigned short sAlo[TM * LDT];
    __shared__ __align__(16) unsigned short sBhi[TN * LDT];
    __shared__ __align__(16) unsigned short sBlo[TN * LDT];

    const int tid  = threadIdx.x;
    const int bm   = blockIdx.x;           // M block
    const int bn   = blockIdx.y;           // N block
    const int wave = tid >> 5;
    const int lane = tid & 31;
    const int wm   = wave >> 2;            // 0..1  (64 rows each)
    const int wn   = wave & 3;             // 0..3  (32 cols each)
    const int lrow  = lane & 15;           // M (A) / N (B) index within 16
    const int khalf = (lane >> 4) << 3;    // 0 or 8: K-half per ISA A/B layout

    const float* xg = x + (size_t)bm * TM * CIN;
    const float* wg = w + (size_t)bn * TN * CIN;

    // Per-thread staging geometry (loop invariant): 4 float4 per tensor per chunk.
    int srow[4], soff[4];
    #pragma unroll
    for (int it = 0; it < 4; ++it) {
        const int idx = tid + (it << 8);   // 0..1023
        srow[it] = idx >> 3;               // 0..127
        const int c4 = idx & 7;            // float4 index within 32-wide K chunk
        soff[it] = srow[it] * LDT + (c4 << 2);
    }

    v8f acc[4][2];
    #pragma unroll
    for (int i = 0; i < 4; ++i)
        #pragma unroll
        for (int j = 0; j < 2; ++j) {
            v8f z = {};
            acc[i][j] = z;
        }

    // Register-staged global loads for the current chunk (issued one iter early).
    float4 ra[4], rb[4];
    #pragma unroll
    for (int it = 0; it < 4; ++it) {
        const int coff = soff[it] - srow[it] * LDT;          // (c4<<2)
        ra[it] = *(const float4*)(xg + (size_t)srow[it] * CIN + coff);
        rb[it] = *(const float4*)(wg + (size_t)srow[it] * CIN + coff);
    }

    for (int k0 = 0; k0 < CIN; k0 += TK) {
        __syncthreads();   // all waves done reading LDS from the previous chunk

        // ---- truncation-split convert + store to LDS ----
        // hi = f truncated to bf16 (top 16 bits), lo = RNE bf16 of (f - hi).
        #pragma unroll
        for (int it = 0; it < 4; ++it) {
            const int off = soff[it];
            float fa[4] = {ra[it].x, ra[it].y, ra[it].z, ra[it].w};
            float fb[4] = {rb[it].x, rb[it].y, rb[it].z, rb[it].w};
            #pragma unroll
            for (int q = 0; q < 4; ++q) {
                const unsigned int ua = __float_as_uint(fa[q]);
                const unsigned int ub = __float_as_uint(fb[q]);
                const float hfa = __uint_as_float(ua & 0xFFFF0000u);
                const float hfb = __uint_as_float(ub & 0xFFFF0000u);
                sAhi[off + q] = (unsigned short)(ua >> 16);   // ds_store_b16_d16_hi path
                sBhi[off + q] = (unsigned short)(ub >> 16);
                sAlo[off + q] = bf16_bits(fa[q] - hfa);       // native RNE cvt
                sBlo[off + q] = bf16_bits(fb[q] - hfb);
            }
        }

        __syncthreads();

        // ---- issue next chunk's global loads early; wmma below hides latency ----
        const int kn = k0 + TK;
        if (kn < CIN) {
            #pragma unroll
            for (int it = 0; it < 4; ++it) {
                const int coff = soff[it] - srow[it] * LDT;
                ra[it] = *(const float4*)(xg + (size_t)srow[it] * CIN + kn + coff);
                rb[it] = *(const float4*)(wg + (size_t)srow[it] * CIN + kn + coff);
            }
            // warm L2 for the chunk after next (global_prefetch_b8)
            if (kn + TK < CIN) {
                __builtin_prefetch(xg + (size_t)srow[0] * CIN + kn + TK, 0, 1);
                __builtin_prefetch(wg + (size_t)srow[0] * CIN + kn + TK, 0, 1);
            }
        }

        // ---- load fragments (two ds_load_b128 per fragment, ISA A/B layout) ----
        v16bf ah[4], al[4], bh[2], bl[2];
        #pragma unroll
        for (int i = 0; i < 4; ++i) {
            const int row = (wm << 6) + (i << 4) + lrow;
            union { v16bf v; v8bf h[2]; } uh, ul;
            uh.h[0] = *(const v8bf*)&sAhi[row * LDT + khalf];
            uh.h[1] = *(const v8bf*)&sAhi[row * LDT + khalf + 16];
            ul.h[0] = *(const v8bf*)&sAlo[row * LDT + khalf];
            ul.h[1] = *(const v8bf*)&sAlo[row * LDT + khalf + 16];
            ah[i] = uh.v;
            al[i] = ul.v;
        }
        #pragma unroll
        for (int j = 0; j < 2; ++j) {
            const int col = (wn << 5) + (j << 4) + lrow;
            union { v16bf v; v8bf h[2]; } uh, ul;
            uh.h[0] = *(const v8bf*)&sBhi[col * LDT + khalf];
            uh.h[1] = *(const v8bf*)&sBhi[col * LDT + khalf + 16];
            ul.h[0] = *(const v8bf*)&sBlo[col * LDT + khalf];
            ul.h[1] = *(const v8bf*)&sBlo[col * LDT + khalf + 16];
            bh[j] = uh.v;
            bl[j] = ul.v;
        }

        // ---- bf16x3 products: hi*hi + lo*hi + hi*lo, f32 accumulate ----
        #pragma unroll
        for (int i = 0; i < 4; ++i)
            #pragma unroll
            for (int j = 0; j < 2; ++j) {
                acc[i][j] = __builtin_amdgcn_wmma_f32_16x16x32_bf16(
                    false, ah[i], false, bh[j], (short)0, acc[i][j], false, false);
                acc[i][j] = __builtin_amdgcn_wmma_f32_16x16x32_bf16(
                    false, al[i], false, bh[j], (short)0, acc[i][j], false, false);
                acc[i][j] = __builtin_amdgcn_wmma_f32_16x16x32_bf16(
                    false, ah[i], false, bl[j], (short)0, acc[i][j], false, false);
            }
    }

    // ---- epilogue: C/D layout (M = vgpr + 8*lane[4], N = lane[3:0]), + bias ----
    #pragma unroll
    for (int j = 0; j < 2; ++j) {
        const int gcol = bn * TN + (wn << 5) + (j << 4) + lrow;
        const float bv = bias[gcol];
        #pragma unroll
        for (int i = 0; i < 4; ++i) {
            const int rbase = bm * TM + (wm << 6) + (i << 4) + ((lane >> 4) << 3);
            #pragma unroll
            for (int v = 0; v < 8; ++v) {
                out[(size_t)(rbase + v) * COUT + gcol] = acc[i][j][v] + bv;
            }
        }
    }
}

extern "C" void kernel_launch(void* const* d_in, const int* in_sizes, int n_in,
                              void* d_out, int out_size, void* d_ws, size_t ws_size,
                              hipStream_t stream) {
    const float* x  = (const float*)d_in[0];
    const float* w  = (const float*)d_in[1];
    const float* b  = (const float*)d_in[2];
    float* out = (float*)d_out;

    dim3 grid(NROWS / TM, COUT / TN);   // 128 x 32 workgroups
    dim3 block(256);                    // 8 waves (wave32)
    sparse_linear_wmma_bf16x3<<<grid, block, 0, stream>>>(x, w, b, out);
}
